// NER_Model_32615981646281
// MI455X (gfx1250) — matrix operand order repaired and checked by reading, not compiled, not run on hardware
//
#include <hip/hip_runtime.h>
#include <hip/hip_bf16.h>

// ---------------------------------------------------------------------------
// NER head, fused pipeline for gfx1250 (MI455X).
//   k1: seg_mean (streaming 100MB read, bf16 compacted output)   [HBM bound]
//   k2: transpose+convert W1 bottom half -> BT[512,768] bf16
//   k3: cls_part[64,512] = out[i,0,:] @ W1_top + b1   (fp32, tiny)
//   k4: h[4096,512] = relu(seg_mean @ W1_bot + cls_part[k/64])   [WMMA bf16]
//       32x64 tile per wave: 8 wmma per 12 b128 loads, ~150MB L2 traffic
//   k5: per-segment logits -> softmax -> -log p[label]           (wave/seg)
//   k6: Npos / pos-sum / bitonic top-M neg-sum -> scalar loss
// ---------------------------------------------------------------------------

typedef __bf16 bf16_t;
typedef __attribute__((ext_vector_type(16))) __bf16 bf16x16;
typedef __attribute__((ext_vector_type(8)))  __bf16 bf16x8;
typedef __attribute__((ext_vector_type(4)))  __bf16 bf16x4;
typedef __attribute__((ext_vector_type(8)))  float  f32x8;

#define BB    64
#define SS    512
#define HH    768
#define TOK   8
#define KSEG  4096
#define N1    512
#define NCLS  4

// --- k1: segment mean (8 contiguous token rows -> 1 row), fp32 -> bf16 -----
__global__ __launch_bounds__(192) void seg_mean_kernel(
    const float* __restrict__ x, bf16_t* __restrict__ A) {
  int k = blockIdx.x;            // segment 0..4095
  int t = threadIdx.x;           // 192 threads x 4 cols = 768
  const float4* base = (const float4*)(x + (size_t)k * TOK * HH);
  float sx = 0.f, sy = 0.f, sz = 0.f, sw = 0.f;
#pragma unroll
  for (int tok = 0; tok < TOK; ++tok) {
    float4 v = base[tok * (HH / 4) + t];
    sx += v.x; sy += v.y; sz += v.z; sw += v.w;
  }
  bf16x4 o;
  o[0] = (bf16_t)(sx * 0.125f);
  o[1] = (bf16_t)(sy * 0.125f);
  o[2] = (bf16_t)(sz * 0.125f);
  o[3] = (bf16_t)(sw * 0.125f);
  *(bf16x4*)(A + (size_t)k * HH + t * 4) = o;
}

// --- k2: BT[n][k] = bf16(W1[768+k][n])  (coalesced read, scattered write) ---
__global__ __launch_bounds__(256) void w1t_kernel(
    const float* __restrict__ W1, bf16_t* __restrict__ BT) {
  int id = blockIdx.x * 256 + threadIdx.x;   // 0 .. 512*768-1
  int n  = id & (N1 - 1);
  int kk = id >> 9;
  BT[(size_t)n * HH + kk] = (bf16_t)W1[(size_t)(HH + kk) * N1 + n];
}

// --- k3: cls_part[i][n] = b1[n] + out[i,0,:] . W1_top[:,n]  (fp32) ---------
__global__ __launch_bounds__(256) void cls_kernel(
    const float* __restrict__ x, const float* __restrict__ W1,
    const float* __restrict__ b1, float* __restrict__ cp) {
  int id = blockIdx.x * 256 + threadIdx.x;   // 0 .. 64*512-1
  int n  = id & (N1 - 1);
  int i  = id >> 9;
  const float* cls = x + (size_t)i * SS * HH;   // row i, token 0
  float acc = b1[n];
  for (int kk = 0; kk < HH; ++kk)
    acc = fmaf(cls[kk], W1[(size_t)kk * N1 + n], acc);
  cp[id] = acc;
}

// --- k4: WMMA GEMM  h = relu(A @ BT^T + cls_part)  -------------------------
// Wave tile 32(M) x 64(N): 2 A-frags x 4 B-frags -> 8 wmma / K-step.
// Block = 8 waves covering a 64x256 super-tile (A rows shared by 4 waves).
// Fragment layouts follow CDNA5 ISA 7.12.2 (16-bit A 16x32, B 32x16, f32 C).
__global__ __launch_bounds__(256) void gemm1_kernel(
    const bf16_t* __restrict__ A, const bf16_t* __restrict__ BT,
    const float* __restrict__ cp, bf16_t* __restrict__ Hb) {
  int wave = threadIdx.x >> 5;
  int lane = threadIdx.x & 31;
  int l16  = lane & 15;
  int kh   = lane >> 4;            // K-half select per ISA layout

  int bm = blockIdx.x >> 1;        // 0..63  (M blocks of 64)
  int bn = blockIdx.x & 1;         // 0..1   (N blocks of 256)
  int mw = wave >> 2;              // 0..1
  int nw = wave & 3;               // 0..3
  int m0 = bm * 64 + mw * 32;      // wave M base (32 rows, inside one cls row)
  int n0 = bn * 256 + nw * 64;     // wave N base (64 cols)

  const bf16_t* aRow0 = A + (size_t)(m0 + l16) * HH;
  const bf16_t* aRow1 = aRow0 + (size_t)16 * HH;
  const bf16_t* bRow[4];
#pragma unroll
  for (int f = 0; f < 4; ++f)
    bRow[f] = BT + (size_t)(n0 + f * 16 + l16) * HH;

  f32x8 c[2][4];
#pragma unroll
  for (int mf = 0; mf < 2; ++mf)
#pragma unroll
    for (int nf = 0; nf < 4; ++nf) c[mf][nf] = (f32x8){};

  for (int kk = 0; kk < HH; kk += 32) {
    union { bf16x16 v; bf16x8 h[2]; } a0, a1, b[4];
    // A 16x32: VGPR0-3 = K kk+8*kh..+7, VGPR4-7 = K kk+16+8*kh..+7
    a0.h[0] = *(const bf16x8*)(aRow0 + kk + kh * 8);
    a0.h[1] = *(const bf16x8*)(aRow0 + kk + 16 + kh * 8);
    a1.h[0] = *(const bf16x8*)(aRow1 + kk + kh * 8);
    a1.h[1] = *(const bf16x8*)(aRow1 + kk + 16 + kh * 8);
    // B 32x16: lane holds col n, K kk+8*kh..+7 then kk+16+8*kh..+7
#pragma unroll
    for (int nf = 0; nf < 4; ++nf) {
      const bf16_t* bp = bRow[nf] + kk + kh * 8;
      b[nf].h[0] = *(const bf16x8*)(bp);
      b[nf].h[1] = *(const bf16x8*)(bp + 16);
    }
#pragma unroll
    for (int nf = 0; nf < 4; ++nf) {
      c[0][nf] = __builtin_amdgcn_wmma_f32_16x16x32_bf16(
          false, a0.v, false, b[nf].v, (short)0, c[0][nf], false, false);
      c[1][nf] = __builtin_amdgcn_wmma_f32_16x16x32_bf16(
          false, a1.v, false, b[nf].v, (short)0, c[1][nf], false, false);
    }
  }
  // Epilogue: + cls_part (batch row constant over the 32-row tile), ReLU.
  int i = m0 >> 6;                 // == bm
#pragma unroll
  for (int nf = 0; nf < 4; ++nf) {
    int ng = n0 + nf * 16 + l16;
    float clsv = cp[i * N1 + ng];
#pragma unroll
    for (int mf = 0; mf < 2; ++mf) {
#pragma unroll
      for (int r = 0; r < 8; ++r) {
        int m = m0 + mf * 16 + r + kh * 8;  // f32 C layout: lane>=16 -> +8
        float v = c[mf][nf][r] + clsv;
        v = v > 0.f ? v : 0.f;
        Hb[(size_t)m * N1 + ng] = (bf16_t)v;
      }
    }
  }
}

// --- k5: logits -> softmax -> cost, one wave per segment -------------------
__global__ __launch_bounds__(256) void logits_kernel(
    const bf16_t* __restrict__ Hb, const float* __restrict__ W2,
    const float* __restrict__ b2v, const int* __restrict__ labels,
    float* __restrict__ costs, int* __restrict__ labs) {
  int wave = threadIdx.x >> 5;
  int lane = threadIdx.x & 31;
  int k = blockIdx.x * 8 + wave;
  const bf16_t* hrow = Hb + (size_t)k * N1;
  float a0 = 0.f, a1 = 0.f, a2 = 0.f, a3 = 0.f;
  for (int j = lane; j < N1; j += 32) {
    float hv = (float)hrow[j];
    float4 wr = *(const float4*)(W2 + j * 4);
    a0 = fmaf(hv, wr.x, a0); a1 = fmaf(hv, wr.y, a1);
    a2 = fmaf(hv, wr.z, a2); a3 = fmaf(hv, wr.w, a3);
  }
#pragma unroll
  for (int off = 16; off > 0; off >>= 1) {
    a0 += __shfl_xor(a0, off, 32);
    a1 += __shfl_xor(a1, off, 32);
    a2 += __shfl_xor(a2, off, 32);
    a3 += __shfl_xor(a3, off, 32);
  }
  if (lane == 0) {
    float l0 = a0 + b2v[0], l1 = a1 + b2v[1];
    float l2 = a2 + b2v[2], l3 = a3 + b2v[3];
    float mx = fmaxf(fmaxf(l0, l1), fmaxf(l2, l3));
    float s = expf(l0 - mx) + expf(l1 - mx) + expf(l2 - mx) + expf(l3 - mx);
    int lab = labels[(size_t)k * TOK];   // first token of segment
    float lsel = (lab == 0) ? l0 : (lab == 1) ? l1 : (lab == 2) ? l2 : l3;
    costs[k] = -(lsel - mx - logf(s));   // -log softmax[label]
    labs[k]  = lab;
  }
}

// --- k6: Npos, pos-sum, exact descending bitonic sort of neg costs, top-M --
__global__ __launch_bounds__(1024) void final_kernel(
    const float* __restrict__ costs, const int* __restrict__ labs,
    float* __restrict__ outp) {
  __shared__ float vals[KSEG];
  __shared__ int   s_npos;
  __shared__ float s_possum, s_negsum;
  int t = threadIdx.x;
  if (t == 0) { s_npos = 0; s_possum = 0.f; s_negsum = 0.f; }
  __syncthreads();
  int lnpos = 0; float lpos = 0.f;
  for (int j = t; j < KSEG; j += 1024) {
    int lab = labs[j]; float c = costs[j];
    if (lab > 0) { lnpos++; lpos += c; vals[j] = -__builtin_inff(); }
    else vals[j] = c;
  }
  atomicAdd(&s_npos, lnpos);
  atomicAdd(&s_possum, lpos);
  __syncthreads();
  // Bitonic sort, descending overall (finite first, -inf last).
  for (int size = 2; size <= KSEG; size <<= 1) {
    for (int stride = size >> 1; stride > 0; stride >>= 1) {
      for (int p = t; p < KSEG / 2; p += 1024) {
        int lo = 2 * p - (p & (stride - 1));
        int hi = lo + stride;
        bool desc = ((lo & size) == 0);
        float a = vals[lo], b = vals[hi];
        bool sw = desc ? (a < b) : (a > b);
        if (sw) { vals[lo] = b; vals[hi] = a; }
      }
      __syncthreads();
    }
  }
  int npos = s_npos;
  int M = (int)floorf(2.5f * (float)npos);
  if (M > KSEG) M = KSEG;
  float lneg = 0.f;
  for (int j = t; j < M; j += 1024) {
    float v = vals[j];
    if (__builtin_isfinite(v)) lneg += v;   // mirrors isfinite mask in ref
  }
  atomicAdd(&s_negsum, lneg);
  __syncthreads();
  if (t == 0) {
    float denom = floorf(3.5f * (float)npos);
    outp[0] = (s_possum + s_negsum) / denom;
  }
}

// ---------------------------------------------------------------------------
extern "C" void kernel_launch(void* const* d_in, const int* in_sizes, int n_in,
                              void* d_out, int out_size, void* d_ws, size_t ws_size,
                              hipStream_t stream) {
  const float* x   = (const float*)d_in[0];   // out [64,512,768]
  const float* W1  = (const float*)d_in[1];   // [1536,512]
  const float* b1  = (const float*)d_in[2];   // [512]
  const float* W2  = (const float*)d_in[3];   // [512,4]
  const float* b2  = (const float*)d_in[4];   // [4]
  const int*   labels = (const int*)d_in[5];  // [64,512]
  // d_in[6] = sent_ids: structurally id = flat_index/8 -> derived, unused.
  (void)in_sizes; (void)n_in; (void)out_size; (void)ws_size;

  char* w = (char*)d_ws;
  bf16_t* A   = (bf16_t*)w;  w += (size_t)KSEG * HH * 2;   // 6,291,456
  bf16_t* BT  = (bf16_t*)w;  w += (size_t)N1 * HH * 2;     //   786,432
  float*  cp  = (float*)w;   w += (size_t)BB * N1 * 4;     //   131,072
  bf16_t* Hb  = (bf16_t*)w;  w += (size_t)KSEG * N1 * 2;   // 4,194,304
  float*  costs = (float*)w; w += (size_t)KSEG * 4;        //    16,384
  int*    labsW = (int*)w;   w += (size_t)KSEG * 4;        //    16,384

  seg_mean_kernel<<<KSEG, 192, 0, stream>>>(x, A);
  w1t_kernel<<<(N1 * HH) / 256, 256, 0, stream>>>(W1, BT);
  cls_kernel<<<(BB * N1) / 256, 256, 0, stream>>>(x, W1, b1, cp);
  gemm1_kernel<<<128, 256, 0, stream>>>(A, BT, cp, Hb);
  logits_kernel<<<512, 256, 0, stream>>>(Hb, W2, b2, labels, costs, labsW);
  final_kernel<<<1, 1024, 0, stream>>>(costs, labsW, (float*)d_out);
}
